// MultiScaleAttention_8650064134347
// MI455X (gfx1250) — compile-verified, matrix operand order checked by reading, not run on hardware
//
#include <hip/hip_runtime.h>
#include <hip/hip_bf16.h>

// MultiScaleAttention for MI455X (gfx1250), wave32 + WMMA f16 (f32 accum).
//   1) qkv_gemm_kernel : x @ qkv_w^T -> q/k/v unpooled [B,H,N,HD] (f16), WMMA
//                        double-buffered LDS, software-pipelined
//   2) pool_ln_kernel  : depthwise 3x3x3 conv (stride 1,2,2) + LayerNorm
//                        (V is written transposed [bh][d][key] for attention)
//   3) attn_kernel     : flash-style attention with WMMA, b128 fragment loads,
//                        K-fragment register double-buffering
//   4) proj_gemm_kernel: attn_out @ proj_w^T + bias -> d_out (f32), WMMA,
//                        A staged via GLOBAL_LOAD_ASYNC_TO_LDS_B128 (ASYNCcnt).
//
// Fragment convention (self-consistent A/B elem<->k mapping):
//   lane_half = (k>>3)&1,  elem(k) = (k&7) + ((k>>4)<<3)
// LDS tiles stored pre-swizzled [tile][lane(32)][elem(16)] halves -> fragment
// load is one aligned v16h (2 x ds_load_b128).

#define BATCH 8
#define HEADS 8
#define HD    96
#define DIMC  768
#define TT    8
#define HH_   28
#define WW_   28
#define NTOK  6273          // 1 + 8*28*28
#define HP    14
#define WP    14
#define NP    1569          // 1 + 8*14*14
#define NPT   1600          // padded key stride for transposed V
#define M_QKV (BATCH * NTOK)   // 50184
#define M_PRJ (BATCH * NP)     // 12552

#if defined(__gfx1250__) &&                                        \
    __has_builtin(__builtin_amdgcn_global_load_async_to_lds_b128) && \
    __has_builtin(__builtin_amdgcn_s_wait_asynccnt)
#define USE_ASYNC_LDS 1
#else
#define USE_ASYNC_LDS 0
#endif

typedef __attribute__((ext_vector_type(16))) _Float16 v16h;
typedef __attribute__((ext_vector_type(8)))  _Float16 v8h;
typedef __attribute__((ext_vector_type(8)))  float    v8f;
#if USE_ASYNC_LDS
typedef int v4i_vs __attribute__((vector_size(16)));
typedef __attribute__((address_space(1))) v4i_vs gv4i;  // global int4
typedef __attribute__((address_space(3))) v4i_vs lv4i;  // LDS int4
#endif

__device__ inline v8f wmma16(v16h a, v16h b, v8f c) {
  return __builtin_amdgcn_wmma_f32_16x16x32_f16(false, a, false, b, (short)0, c,
                                                false, false);
}

__device__ inline v16h cat8(v8h lo, v8h hi) {
  return __builtin_shufflevector(lo, hi, 0, 1, 2, 3, 4, 5, 6, 7, 8, 9, 10, 11,
                                 12, 13, 14, 15);
}

__device__ inline v8h pack8(float4 a, float4 b) {
  v8h h;
  h[0] = (_Float16)a.x; h[1] = (_Float16)a.y;
  h[2] = (_Float16)a.z; h[3] = (_Float16)a.w;
  h[4] = (_Float16)b.x; h[5] = (_Float16)b.y;
  h[6] = (_Float16)b.z; h[7] = (_Float16)b.w;
  return h;
}

// ---------------------------------------------------------------------------
// Kernel 1: qkv = x @ qkv_w^T, scattered into q/k/v [B,HEADS,NTOK,HD] f16.
// Block tile 128(M) x 64(N), K-step 32, 8 waves (4x2), double-buffered LDS.
// ---------------------------------------------------------------------------
__global__ __launch_bounds__(256) void qkv_gemm_kernel(
    const float* __restrict__ x, const float* __restrict__ w,
    _Float16* __restrict__ qun, _Float16* __restrict__ kun,
    _Float16* __restrict__ vun) {
  __shared__ __align__(32) _Float16 smA[2][8 * 512];
  __shared__ __align__(32) _Float16 smB[2][4 * 512];

  const int m0 = blockIdx.x * 128;
  const int n0 = blockIdx.y * 64;
  const int tid  = threadIdx.x;
  const int lane = tid & 31;
  const int wave = tid >> 5;
  const int wm = wave >> 1;
  const int wn = wave & 1;

  v8f acc[2][2];
  v8f z = {};
#pragma unroll
  for (int i = 0; i < 2; ++i)
#pragma unroll
    for (int j = 0; j < 2; ++j) acc[i][j] = z;

  const int ar = tid >> 1;
  const int ac = (tid & 1) << 4;
  const bool aok = (m0 + ar) < M_QKV;
  const float* xrow = x + (size_t)(aok ? (m0 + ar) : 0) * DIMC + ac;
  const int bj = tid & 63;
  const int bkl = (tid >> 6) << 3;
  const float* wrow = w + (size_t)(n0 + bj) * DIMC + bkl;

  const int atile = ar >> 4;
  const int alane = ar & 15;
  const int aeb = (ac >> 1) & 8;
  const int btile = bj >> 4;
  const int blaneF = (bj & 15) + (((bkl >> 3) & 1) << 4);
  const int beb = (bkl >> 1) & 8;

  auto fetchA = [&](int k0, v8h& h0, v8h& h1) {
    if (aok) {
      float4 f0 = *(const float4*)(xrow + k0);
      float4 f1 = *(const float4*)(xrow + k0 + 4);
      float4 f2 = *(const float4*)(xrow + k0 + 8);
      float4 f3 = *(const float4*)(xrow + k0 + 12);
      h0 = pack8(f0, f1);
      h1 = pack8(f2, f3);
    } else {
      v8h zz = {};
      h0 = zz; h1 = zz;
    }
  };
  auto fetchB = [&](int k0) -> v8h {
    float4 g0 = *(const float4*)(wrow + k0);
    float4 g1 = *(const float4*)(wrow + k0 + 4);
    return pack8(g0, g1);
  };
  auto storeA = [&](_Float16* buf, v8h h0, v8h h1) {
    *(v8h*)(buf + atile * 512 + alane * 16 + aeb) = h0;
    *(v8h*)(buf + atile * 512 + (alane + 16) * 16 + aeb) = h1;
  };
  auto storeB = [&](_Float16* buf, v8h hb) {
    *(v8h*)(buf + btile * 512 + blaneF * 16 + beb) = hb;
  };

  // prologue: tile 0 -> buffer 0
  {
    v8h a0, a1;
    fetchA(0, a0, a1);
    v8h hb = fetchB(0);
    storeA(smA[0], a0, a1);
    storeB(smB[0], hb);
  }

  for (int kk = 0; kk < 24; ++kk) {
    const int cur = kk & 1;
    const bool more = kk < 23;
    __syncthreads();

    v8h na0 = {}, na1 = {}, nb = {};
    if (more) {
      const int kn = (kk + 1) << 5;
      if (kk < 22) {  // prefetch k+2 sources
        __builtin_prefetch(xrow + kn + 32, 0, 1);
        __builtin_prefetch(wrow + kn + 32, 0, 1);
      }
      fetchA(kn, na0, na1);
      nb = fetchB(kn);
    }

    const _Float16* A = smA[cur];
    const _Float16* Bm = smB[cur];
#pragma unroll
    for (int nt = 0; nt < 2; ++nt) {
      v16h fb = *(const v16h*)(Bm + (wn * 2 + nt) * 512 + lane * 16);
#pragma unroll
      for (int mt = 0; mt < 2; ++mt) {
        v16h fa = *(const v16h*)(A + (wm * 2 + mt) * 512 + lane * 16);
        acc[mt][nt] = wmma16(fa, fb, acc[mt][nt]);
      }
    }

    if (more) {
      storeA(smA[cur ^ 1], na0, na1);
      storeB(smB[cur ^ 1], nb);
    }
  }

  // epilogue: scatter into q/k/v unpooled f16 [B,HEADS,NTOK,HD]
  const int rowoff = (lane >> 4) << 3;
#pragma unroll
  for (int mt = 0; mt < 2; ++mt) {
#pragma unroll
    for (int nt = 0; nt < 2; ++nt) {
      int jg  = n0 + wn * 32 + nt * 16 + (lane & 15);
      int s   = jg / DIMC;
      int rem = jg - s * DIMC;
      int hh  = rem / HD;
      int dd  = rem - hh * HD;
      _Float16* outp = (s == 0) ? qun : ((s == 1) ? kun : vun);
#pragma unroll
      for (int r = 0; r < 8; ++r) {
        int m = m0 + wm * 32 + mt * 16 + rowoff + r;
        if (m < M_QKV) {
          int b = m / NTOK, n = m - b * NTOK;
          outp[((size_t)(b * HEADS + hh) * NTOK + n) * HD + dd] =
              (_Float16)acc[mt][nt][r];
        }
      }
    }
  }
}

// ---------------------------------------------------------------------------
// Kernel 2: depthwise 3x3x3 conv (stride 1,2,2, pad 1) + cls passthrough + LN.
// transposed!=0: write [bh][d][key] with key-stride NPT (padding zeroed).
// ---------------------------------------------------------------------------
__global__ __launch_bounds__(96) void pool_ln_kernel(
    const _Float16* __restrict__ in, const float* __restrict__ cw,
    const float* __restrict__ lw, const float* __restrict__ lb,
    float* __restrict__ outf, _Float16* __restrict__ outh, int transposed) {
  const int dd    = threadIdx.x;
  const int rowid = blockIdx.x;
  const int npr = transposed ? NPT : NP;
  const int bh = rowid / npr;
  const int np = rowid - bh * npr;

  if (transposed && np >= NP) {
    outh[((size_t)bh * HD + dd) * NPT + np] = (_Float16)0.f;
    return;
  }

  const _Float16* base = in + (size_t)bh * NTOK * HD;
  float val;
  if (np == 0) {
    val = (float)base[dd];
  } else {
    int sp = np - 1;
    int tp = sp / (HP * WP);
    int r2 = sp - tp * (HP * WP);
    int yp = r2 / WP;
    int xp = r2 - yp * WP;
    float acc = 0.f;
#pragma unroll
    for (int dt = 0; dt < 3; ++dt) {
      int t = tp + dt - 1;
      if (t < 0 || t >= TT) continue;
#pragma unroll
      for (int dy = 0; dy < 3; ++dy) {
        int y = 2 * yp + dy - 1;
        if (y < 0 || y >= HH_) continue;
#pragma unroll
        for (int dx = 0; dx < 3; ++dx) {
          int xx = 2 * xp + dx - 1;
          if (xx < 0 || xx >= WW_) continue;
          int n = 1 + (t * HH_ + y) * WW_ + xx;
          acc += cw[((dd * 3 + dt) * 3 + dy) * 3 + dx] *
                 (float)base[(size_t)n * HD + dd];
        }
      }
    }
    val = acc;
  }

  __shared__ float sb[HD];
  sb[dd] = val;
  __syncthreads();
  float s = 0.f, q = 0.f;
#pragma unroll 8
  for (int i = 0; i < HD; ++i) {
    float v = sb[i];
    s += v;
    q += v * v;
  }
  float mean = s * (1.f / HD);
  float var  = q * (1.f / HD) - mean * mean;
  float y = (val - mean) * rsqrtf(var + 1e-5f) * lw[dd] + lb[dd];

  if (outf) outf[((size_t)bh * NP + np) * HD + dd] = y;
  if (transposed)
    outh[((size_t)bh * HD + dd) * NPT + np] = (_Float16)y;
  else
    outh[((size_t)bh * NP + np) * HD + dd] = (_Float16)y;
}

// ---------------------------------------------------------------------------
// Kernel 3: flash attention, one wave per (b,h, 16-row Q tile).
// K fragments register-double-buffered; V rows prefetched one tile ahead.
// ---------------------------------------------------------------------------
__global__ __launch_bounds__(32) void attn_kernel(
    const _Float16* __restrict__ qph, const _Float16* __restrict__ kph,
    const _Float16* __restrict__ vpt, const float* __restrict__ qpf,
    _Float16* __restrict__ oh) {
  const float scale = 0.102062072616f;  // 96^-0.5
  const int qb = blockIdx.x * 16;
  const int hh = blockIdx.y;
  const int bb = blockIdx.z;
  const int bh = bb * HEADS + hh;
  const int lane = threadIdx.x & 31;
  const int n  = lane & 15;
  const int kb = (lane >> 4) << 3;
  const int rowoff = (lane >> 4) << 3;

  // Q A-fragments (3 chunks of K=32 over HD=96), row-clamped.
  v16h qa[3];
  {
    int qrow = qb + n; if (qrow >= NP) qrow = NP - 1;
    const _Float16* qp = qph + ((size_t)bh * NP + qrow) * HD + kb;
#pragma unroll
    for (int c = 0; c < 3; ++c)
      qa[c] = cat8(*(const v8h*)(qp + c * 32), *(const v8h*)(qp + c * 32 + 16));
  }

  auto load_kf = [&](int key0, v16h* b0, v16h* b1) {
    int keyA = key0 + n;      if (keyA >= NP) keyA = NP - 1;
    int keyB = key0 + 16 + n; if (keyB >= NP) keyB = NP - 1;
    const _Float16* ka = kph + ((size_t)bh * NP + keyA) * HD + kb;
    const _Float16* kc = kph + ((size_t)bh * NP + keyB) * HD + kb;
#pragma unroll
    for (int c = 0; c < 3; ++c) {
      b0[c] = cat8(*(const v8h*)(ka + c * 32), *(const v8h*)(ka + c * 32 + 16));
      b1[c] = cat8(*(const v8h*)(kc + c * 32), *(const v8h*)(kc + c * 32 + 16));
    }
  };

  v8f o[6];
  v8f z = {};
#pragma unroll
  for (int j = 0; j < 6; ++j) o[j] = z;
  float mrow[8], lrow[8];
#pragma unroll
  for (int i = 0; i < 8; ++i) { mrow[i] = -1e30f; lrow[i] = 0.f; }

  __shared__ __align__(32) _Float16 sP[512];  // swizzled A-frag layout

  v16h kf0[3], kf1[3], nf0[3], nf1[3];
  load_kf(0, kf0, kf1);

  for (int kt = 0; kt < 50; ++kt) {
    const int key0 = kt * 32;
    const bool more = kt < 49;

    if (more) {
      load_kf(key0 + 32, nf0, nf1);  // next K tile in flight over this tile
#pragma unroll
      for (int j = 0; j < 6; ++j)    // prefetch next V tile rows
        __builtin_prefetch(
            vpt + ((size_t)bh * HD + j * 16 + n) * NPT + key0 + 32 + kb, 0, 1);
    }

    // ---- scores S[16x32] = Q @ K^T ----
    v8f s0 = z, s1 = z;
#pragma unroll
    for (int c = 0; c < 3; ++c) {
      s0 = wmma16(qa[c], kf0[c], s0);
      s1 = wmma16(qa[c], kf1[c], s1);
    }

    // ---- online softmax ----
    bool v0 = (key0 + n) < NP;
    bool v1 = (key0 + 16 + n) < NP;
    float p0[8], p1[8], alpha[8];
#pragma unroll
    for (int rr = 0; rr < 8; ++rr) {
      float a0 = v0 ? s0[rr] * scale : -1e30f;
      float a1 = v1 ? s1[rr] * scale : -1e30f;
      float t = fmaxf(a0, a1);
      t = fmaxf(t, __shfl_xor(t, 1, 32));
      t = fmaxf(t, __shfl_xor(t, 2, 32));
      t = fmaxf(t, __shfl_xor(t, 4, 32));
      t = fmaxf(t, __shfl_xor(t, 8, 32));
      float mn = fmaxf(mrow[rr], t);
      float al = __expf(mrow[rr] - mn);
      float e0 = __expf(a0 - mn);
      float e1 = __expf(a1 - mn);
      float rs = e0 + e1;
      rs += __shfl_xor(rs, 1, 32);
      rs += __shfl_xor(rs, 2, 32);
      rs += __shfl_xor(rs, 4, 32);
      rs += __shfl_xor(rs, 8, 32);
      lrow[rr] = lrow[rr] * al + rs;
      mrow[rr] = mn;
      alpha[rr] = al;
      p0[rr] = e0;
      p1[rr] = e1;
    }
#pragma unroll
    for (int j = 0; j < 6; ++j)
#pragma unroll
      for (int rr = 0; rr < 8; ++rr) o[j][rr] *= alpha[rr];

    // ---- P into swizzled A-frag layout ----
    __syncthreads();
#pragma unroll
    for (int rr = 0; rr < 8; ++rr) {
      int lf = (rowoff + rr) + (((lane >> 3) & 1) << 4);
      sP[lf * 16 + (lane & 7)] = (_Float16)p0[rr];
      sP[lf * 16 + 8 + (lane & 7)] = (_Float16)p1[rr];
    }
    __syncthreads();
    v16h pa = *(const v16h*)(sP + lane * 16);

    // ---- O += P @ V ----
#pragma unroll
    for (int j = 0; j < 6; ++j) {
      const _Float16* vp =
          vpt + ((size_t)bh * HD + j * 16 + n) * NPT + key0 + kb;
      v16h bv = cat8(*(const v8h*)vp, *(const v8h*)(vp + 16));
      o[j] = wmma16(pa, bv, o[j]);
    }

    if (more) {
#pragma unroll
      for (int c = 0; c < 3; ++c) { kf0[c] = nf0[c]; kf1[c] = nf1[c]; }
    }
  }

  // ---- epilogue: O/l + residual q ----
#pragma unroll
  for (int rr = 0; rr < 8; ++rr) {
    int rowg = qb + rowoff + rr;
    if (rowg < NP) {
      float inv = 1.f / lrow[rr];
#pragma unroll
      for (int j = 0; j < 6; ++j) {
        int dd = j * 16 + n;
        float val = o[j][rr] * inv + qpf[((size_t)bh * NP + rowg) * HD + dd];
        oh[((size_t)bb * NP + rowg) * DIMC + hh * HD + dd] = (_Float16)val;
      }
    }
  }
}

// ---------------------------------------------------------------------------
// Kernel 4: out = o @ proj_w^T + proj_b (f32). Double-buffered; A staged via
// async global->LDS copies when available (ASYNCcnt path).
// ---------------------------------------------------------------------------
__global__ __launch_bounds__(256) void proj_gemm_kernel(
    const _Float16* __restrict__ a, const float* __restrict__ w,
    const float* __restrict__ bias, float* __restrict__ out) {
  __shared__ __align__(32) _Float16 smA[2][8 * 512];
  __shared__ __align__(32) _Float16 smB[2][4 * 512];

  const int m0 = blockIdx.x * 128;
  const int n0 = blockIdx.y * 64;
  const int tid  = threadIdx.x;
  const int lane = tid & 31;
  const int wave = tid >> 5;
  const int wm = wave >> 1;
  const int wn = wave & 1;

  v8f acc[2][2];
  v8f z = {};
#pragma unroll
  for (int i = 0; i < 2; ++i)
#pragma unroll
    for (int j = 0; j < 2; ++j) acc[i][j] = z;

  const int ar = tid >> 1;
  const int ac = (tid & 1) << 4;
  int arow_i = m0 + ar; if (arow_i >= M_PRJ) arow_i = M_PRJ - 1;  // clamp
  const _Float16* arow = a + (size_t)arow_i * DIMC + ac;
  const int bj = tid & 63;
  const int bkl = (tid >> 6) << 3;
  const float* wrow = w + (size_t)(n0 + bj) * DIMC + bkl;

  const int atile = ar >> 4;
  const int alane = ar & 15;
  const int aeb = (ac >> 1) & 8;
  const int btile = bj >> 4;
  const int blaneF = (bj & 15) + (((bkl >> 3) & 1) << 4);
  const int beb = (bkl >> 1) & 8;

  auto issueA = [&](int k0, _Float16* buf) {
#if USE_ASYNC_LDS
    __builtin_amdgcn_global_load_async_to_lds_b128(
        (gv4i*)(arow + k0),
        (lv4i*)(buf + atile * 512 + alane * 16 + aeb), 0, 0);
    __builtin_amdgcn_global_load_async_to_lds_b128(
        (gv4i*)(arow + k0 + 8),
        (lv4i*)(buf + atile * 512 + (alane + 16) * 16 + aeb), 0, 0);
#else
    v8h h0 = *(const v8h*)(arow + k0);
    v8h h1 = *(const v8h*)(arow + k0 + 8);
    *(v8h*)(buf + atile * 512 + alane * 16 + aeb) = h0;
    *(v8h*)(buf + atile * 512 + (alane + 16) * 16 + aeb) = h1;
#endif
  };
  auto fetchB = [&](int k0) -> v8h {
    float4 g0 = *(const float4*)(wrow + k0);
    float4 g1 = *(const float4*)(wrow + k0 + 4);
    return pack8(g0, g1);
  };
  auto storeB = [&](_Float16* buf, v8h hb) {
    *(v8h*)(buf + btile * 512 + blaneF * 16 + beb) = hb;
  };

  // prologue: tile 0 -> buffer 0
  issueA(0, smA[0]);
  storeB(smB[0], fetchB(0));

  for (int kk = 0; kk < 24; ++kk) {
    const int cur = kk & 1;
    const bool more = kk < 23;
#if USE_ASYNC_LDS
    __builtin_amdgcn_s_wait_asynccnt(0);
#endif
    __syncthreads();

    v8h nb = {};
    if (more) {
      const int kn = (kk + 1) << 5;
      issueA(kn, smA[cur ^ 1]);  // async copy lands in the other buffer
      nb = fetchB(kn);
      if (kk < 22) __builtin_prefetch(wrow + kn + 32, 0, 1);
    }

    const _Float16* A = smA[cur];
    const _Float16* Bm = smB[cur];
#pragma unroll
    for (int nt = 0; nt < 2; ++nt) {
      v16h fb = *(const v16h*)(Bm + (wn * 2 + nt) * 512 + lane * 16);
#pragma unroll
      for (int mt = 0; mt < 2; ++mt) {
        v16h fa = *(const v16h*)(A + (wm * 2 + mt) * 512 + lane * 16);
        acc[mt][nt] = wmma16(fa, fb, acc[mt][nt]);
      }
    }

    if (more) storeB(smB[cur ^ 1], nb);
  }

  const int rowoff = (lane >> 4) << 3;
#pragma unroll
  for (int mt = 0; mt < 2; ++mt) {
#pragma unroll
    for (int nt = 0; nt < 2; ++nt) {
      int jg = n0 + wn * 32 + nt * 16 + (lane & 15);
      float bv = bias[jg];
#pragma unroll
      for (int r = 0; r < 8; ++r) {
        int m = m0 + wm * 32 + mt * 16 + rowoff + r;
        if (m < M_PRJ) out[(size_t)m * DIMC + jg] = acc[mt][nt][r] + bv;
      }
    }
  }
}

// ---------------------------------------------------------------------------
extern "C" void kernel_launch(void* const* d_in, const int* in_sizes, int n_in,
                              void* d_out, int out_size, void* d_ws,
                              size_t ws_size, hipStream_t stream) {
  const float* x     = (const float*)d_in[0];
  const float* qkvw  = (const float*)d_in[1];
  const float* pqw   = (const float*)d_in[2];
  const float* pkw   = (const float*)d_in[3];
  const float* pvw   = (const float*)d_in[4];
  const float* nqw   = (const float*)d_in[5];
  const float* nqb   = (const float*)d_in[6];
  const float* nkw   = (const float*)d_in[7];
  const float* nkb   = (const float*)d_in[8];
  const float* nvw   = (const float*)d_in[9];
  const float* nvb   = (const float*)d_in[10];
  const float* projw = (const float*)d_in[11];
  const float* projb = (const float*)d_in[12];
  float* out = (float*)d_out;

  char* ws = (char*)d_ws;
  size_t off = 0;
  auto carve = [&](size_t bytes) -> void* {
    void* p = ws + off;
    off = (off + bytes + 255) & ~(size_t)255;
    return p;
  };
  const size_t UN  = (size_t)BATCH * HEADS * NTOK * HD;
  const size_t PN  = (size_t)BATCH * HEADS * NP * HD;
  const size_t PNT = (size_t)BATCH * HEADS * HD * NPT;
  _Float16* qun = (_Float16*)carve(UN * 2);
  _Float16* kun = (_Float16*)carve(UN * 2);
  _Float16* vun = (_Float16*)carve(UN * 2);
  float*    qpf = (float*)carve(PN * 4);
  _Float16* qph = (_Float16*)carve(PN * 2);
  _Float16* kph = (_Float16*)carve(PN * 2);
  _Float16* vpt = (_Float16*)carve(PNT * 2);  // V transposed [bh][d][key]
  _Float16* oh  = (_Float16*)carve(PN * 2);   // [B,NP,DIM] head-interleaved

  qkv_gemm_kernel<<<dim3((M_QKV + 127) / 128, (3 * DIMC) / 64), 256, 0,
                    stream>>>(x, qkvw, qun, kun, vun);

  const int rows  = BATCH * HEADS * NP;
  const int rowsT = BATCH * HEADS * NPT;
  pool_ln_kernel<<<dim3(rows), 96, 0, stream>>>(qun, pqw, nqw, nqb, qpf, qph, 0);
  pool_ln_kernel<<<dim3(rows), 96, 0, stream>>>(kun, pkw, nkw, nkb, nullptr, kph, 0);
  pool_ln_kernel<<<dim3(rowsT), 96, 0, stream>>>(vun, pvw, nvw, nvb, nullptr, vpt, 1);

  attn_kernel<<<dim3((NP + 15) / 16, HEADS, BATCH), 32, 0, stream>>>(
      qph, kph, vpt, qpf, oh);

  proj_gemm_kernel<<<dim3((M_PRJ + 127) / 128, DIMC / 64), 256, 0, stream>>>(
      oh, projw, projb, out);
}
